// MultiGranularitySemanticModule_48713519071711
// MI455X (gfx1250) — matrix operand order repaired and checked by reading, not compiled, not run on hardware
//
#include <hip/hip_runtime.h>

// ---------------- problem constants ----------------
#define V_  20000
#define M_  32
#define E_  128
#define H_  4
#define N_  8192
#define L_  64
#define C_  128

typedef __bf16 bf16;
typedef __attribute__((ext_vector_type(16))) bf16  v16bf;
typedef __attribute__((ext_vector_type(8)))  bf16  v8bf;
typedef __attribute__((ext_vector_type(8)))  float v8f;

// D = A(16x32 bf16) * B(32x16 bf16) + C(16x16 f32)
__device__ inline v8f wmma_bf16(v16bf a, v16bf b, v8f c) {
  return __builtin_amdgcn_wmma_f32_16x16x32_bf16(false, a, false, b, (short)0, c,
                                                 false, false);
}

// 32-bit LDS byte address of a shared-memory pointer (for async-to-LDS VDST).
__device__ inline unsigned lds_off(const void* p) {
  return (unsigned)(unsigned long long)
      (__attribute__((address_space(3))) const char*)(const char*)p;
}

// A-fragment (16x32 bf16, row-major in LDS, ld = row stride in elements).
// Per ISA 7.12.2: lane m=lane&15, half=lane>>4; VGPR j<4 -> K = half*8+2j(+1);
// VGPR j>=4 -> K = 16 + half*8 + 2(j-4)(+1). Both halves are 8 contiguous bf16.
__device__ inline v16bf lds_loadA(const bf16* A, int ld, int row, int kbase, int lane) {
  const int half = (lane >> 4) & 1;
  const bf16* p = A + row * ld + kbase + half * 8;
  v8bf lo = *(const v8bf*)(p);
  v8bf hi = *(const v8bf*)(p + 16);
  v16bf a;
#pragma unroll
  for (int i = 0; i < 8; ++i) { a[i] = lo[i]; a[8 + i] = hi[i]; }
  return a;
}

// B-fragment from prepacked buffer: one 32B contiguous load per lane.
__device__ inline v16bf loadB(const bf16* pack, int ktileLin, int ntile, int lane) {
  return *(const v16bf*)(pack + (((ktileLin * 8 + ntile) * 32) + lane) * 16);
}

// ------------- pack B matrices into fragment order -------------
// B(k,n) = src[offset + (k>>kmodShift)*strideDt + (k & mask)*strideK + n*strideN]
// dst layout: [ktile][ntile=8][lane=32][16 bf16], lane holds n = ntile*16+(lane&15),
// half = lane>>4: j<4 -> K = ktile*32 + half*8 + 2j (+1); j>=4 -> K = ktile*32 + 16 + half*8 + 2(j-4) (+1).
__global__ __launch_bounds__(128) void pack_b_frags(const float* __restrict__ src,
                                                    bf16* __restrict__ dst,
                                                    int KTtot, int strideK, int strideN,
                                                    int offset, int kmodShift, int strideDt) {
  int t = blockIdx.x * 128 + threadIdx.x;
  if (t >= KTtot * 8 * 32) return;
  int lane  = t & 31;
  int ntile = (t >> 5) & 7;
  int ktile = t >> 8;
  int nn   = ntile * 16 + (lane & 15);
  int half = (lane >> 4) & 1;
  int mask = (1 << kmodShift) - 1;
  bf16* d = dst + t * 16;
#pragma unroll
  for (int j = 0; j < 8; ++j) {
    int k0 = ktile * 32 + (j < 4 ? half * 8 + 2 * j : 16 + half * 8 + 2 * (j - 4));
    int e  = k0 & mask;
    int dt = k0 >> kmodShift;
    long idx = (long)offset + (long)dt * strideDt + (long)e * strideK + (long)nn * strideN;
    d[2 * j]     = (bf16)src[idx];
    d[2 * j + 1] = (bf16)src[idx + strideK];  // k0 even -> same dt, e+1
  }
}

// ------------- stage 1: per-word masked-mean query + 1-query MHA -------------
__global__ __launch_bounds__(128) void word_attn_kernel(
    const int* __restrict__ word2news, const int* __restrict__ w2n_len,
    const float* __restrict__ table, const float* __restrict__ in_w,
    const float* __restrict__ in_b, const float* __restrict__ out_w,
    const float* __restrict__ out_b, const bf16* __restrict__ wk_pack,
    const bf16* __restrict__ wv_pack, bf16* __restrict__ we) {
  const int v = blockIdx.x;
  const int tid = threadIdx.x;

  __shared__ float ctx_f[M_ * E_];
  __shared__ bf16  ctx_b[M_ * E_];
  __shared__ float kh[M_ * E_];
  __shared__ float vh[M_ * E_];
  __shared__ float qh[E_];
  __shared__ float attn[H_ * M_];
  __shared__ float osh[E_];

  const int s = w2n_len[v];
  if (s == 0) {                 // block-uniform branch: words with no news -> 0
    we[v * E_ + tid] = (bf16)0.0f;
    return;
  }

  { // gather ctx rows (32 x 128 f32), keep f32 + bf16 copies
    int m = tid >> 2, q4 = tid & 3;
    int nidx = word2news[v * M_ + m];
    const float* src = table + (long)nidx * E_ + q4 * 32;
    float* dstf = ctx_f + m * E_ + q4 * 32;
    bf16*  dstb = ctx_b + m * E_ + q4 * 32;
#pragma unroll 8
    for (int i = 0; i < 32; ++i) { float x = src[i]; dstf[i] = x; dstb[i] = (bf16)x; }
  }
  __syncthreads();

  // masked mean query (thread = embedding dim)
  float qv = 0.0f;
  for (int m = 0; m < s; ++m) qv += ctx_f[m * E_ + tid];
  qv /= (float)s;
  qh[tid] = qv;
  __syncthreads();

  // q projection: qh_proj[e] = bq[e] + sum_j q[j] * wq[e][j]
  float qproj = in_b[tid];
  for (int j = 0; j < E_; ++j) qproj += qh[j] * in_w[tid * E_ + j];
  __syncthreads();
  qh[tid] = qproj;   // consumed only after the post-GEMM barrier

  // K and V projections: [32,128] @ [128,128] via bf16 WMMA, f32 accumulate
  const int lane = tid & 31, wid = tid >> 5;
  for (int t = wid; t < 16; t += 4) {
    int mtile = t >> 3, ntile = t & 7;
    v8f acck = {}, accv = {};
#pragma unroll
    for (int kt = 0; kt < 4; ++kt) {
      v16bf a  = lds_loadA(ctx_b, E_, mtile * 16 + (lane & 15), kt * 32, lane);
      v16bf bk = loadB(wk_pack, kt, ntile, lane);
      v16bf bv = loadB(wv_pack, kt, ntile, lane);
      acck = wmma_bf16(a, bk, acck);
      accv = wmma_bf16(a, bv, accv);
    }
    int col  = ntile * 16 + (lane & 15);
    int row0 = mtile * 16 + ((lane >> 4) << 3);
    float bkb = in_b[E_ + col], bvb = in_b[2 * E_ + col];
#pragma unroll
    for (int r = 0; r < 8; ++r) {
      kh[(row0 + r) * E_ + col] = acck[r] + bkb;
      vh[(row0 + r) * E_ + col] = accv[r] + bvb;
    }
  }
  __syncthreads();

  // scores: one wave per head (wave32), lane = news slot m
  float sc;
  {
    int h = wid, m = lane;
    float d = 0.0f;
#pragma unroll 8
    for (int j = 0; j < 32; ++j) d += qh[h * 32 + j] * kh[m * E_ + h * 32 + j];
    sc = d * 0.17677669529663687f;  // 1/sqrt(Dh=32)
    if (m >= s) sc = -1e9f;
  }
  // wave-level softmax
  float mx = sc;
  for (int o = 16; o > 0; o >>= 1) mx = fmaxf(mx, __shfl_xor(mx, o, 32));
  float ex = __expf(sc - mx);
  float sm = ex;
  for (int o = 16; o > 0; o >>= 1) sm += __shfl_xor(sm, o, 32);
  attn[tid] = ex / sm;
  __syncthreads();

  // o[h,d] = sum_m attn[h,m] * vh[m, h*32+d]
  {
    int h = wid, dd = lane;
    float acc = 0.0f;
#pragma unroll 8
    for (int m = 0; m < 32; ++m) acc += attn[h * 32 + m] * vh[m * E_ + h * 32 + dd];
    osh[h * 32 + dd] = acc;
  }
  __syncthreads();

  // output projection -> bf16 word embedding
  float r = out_b[tid];
  for (int j = 0; j < E_; ++j) r += osh[j] * out_w[tid * E_ + j];
  we[v * E_ + tid] = (bf16)r;
}

// ------------- stage 2: implicit-GEMM conv + relu + maxpool + FC -------------
template <int KS>
__device__ inline void conv_one(const bf16* A, const bf16* __restrict__ pack,
                                const float* __restrict__ bias, unsigned* pool,
                                int lane, int wid) {
  const int T = L_ - KS + 1;
  for (int t = wid; t < 32; t += 4) {
    int mtile = t >> 3, ntile = t & 7;
    v8f acc = {};
#pragma unroll
    for (int dt = 0; dt < KS; ++dt)
#pragma unroll
      for (int kt = 0; kt < 4; ++kt) {
        v16bf a = lds_loadA(A, E_, mtile * 16 + (lane & 15) + dt, kt * 32, lane);
        v16bf b = loadB(pack, dt * 4 + kt, ntile, lane);
        acc = wmma_bf16(a, b, acc);
      }
    int col  = ntile * 16 + (lane & 15);
    int row0 = mtile * 16 + ((lane >> 4) << 3);
    float b0 = bias[col];
#pragma unroll
    for (int r = 0; r < 8; ++r) {
      int rr = row0 + r;
      if (rr < T) {
        float vv = acc[r] + b0;
        vv = vv > 0.0f ? vv : 0.0f;   // relu: non-negative -> uint-max == float-max
        atomicMax(&pool[col], __float_as_uint(vv));
      }
    }
  }
}

__global__ __launch_bounds__(128) void conv_fc_kernel(
    const int* __restrict__ news_words, const bf16* __restrict__ we,
    const bf16* __restrict__ cp3, const float* __restrict__ cb3,
    const bf16* __restrict__ cp4, const float* __restrict__ cb4,
    const bf16* __restrict__ cp5, const float* __restrict__ cb5,
    const float* __restrict__ fc_w, const float* __restrict__ fc_b,
    float* __restrict__ out) {
  const int n = blockIdx.x, tid = threadIdx.x;
  __shared__ bf16 A[(L_ + 4) * E_];       // 4 zero-pad rows for tap shifts
  __shared__ unsigned pooled[3 * C_];     // float bits, init 0 (== 0.0f)

  { // gather doc rows with the CDNA5 async global->LDS path (ASYNCcnt-tracked):
    // 16 lanes per row, 16B per lane, 8 rows per issue round.
    const unsigned ldsA = lds_off(A);
    const int rowblk = tid >> 4;          // 0..7
    const int seg    = tid & 15;          // 16-byte segment within a row
#pragma unroll
    for (int it = 0; it < 8; ++it) {
      int row = it * 8 + rowblk;
      int w = news_words[n * L_ + row];
      unsigned goff  = (unsigned)w * (E_ * 2) + seg * 16;       // bytes into we
      unsigned laddr = ldsA + (unsigned)row * (E_ * 2) + seg * 16;
      asm volatile("global_load_async_to_lds_b128 %0, %1, %2"
                   :: "v"(laddr), "v"(goff), "s"(we) : "memory");
    }
  }
  // overlap with the async copies: zero pooled + pad rows, warm fc_w row
  for (int i = tid; i < 3 * C_; i += 128) pooled[i] = 0u;
  for (int i = tid; i < 4 * E_; i += 128) A[L_ * E_ + i] = (bf16)0.0f;
  __builtin_prefetch(fc_w + tid * (3 * C_), 0, 3);   // global_prefetch_b8
  asm volatile("s_wait_asynccnt 0" ::: "memory");
  __syncthreads();

  const int lane = tid & 31, wid = tid >> 5;
  conv_one<3>(A, cp3, cb3, pooled + 0 * C_, lane, wid);
  conv_one<4>(A, cp4, cb4, pooled + 1 * C_, lane, wid);
  conv_one<5>(A, cp5, cb5, pooled + 2 * C_, lane, wid);
  __syncthreads();

  // FC: out[n][e] = fc_b[e] + feats . fc_w[e,:]
  float r = fc_b[tid];
  for (int j = 0; j < 3 * C_; ++j)
    r += __uint_as_float(pooled[j]) * fc_w[tid * (3 * C_) + j];
  out[(long)n * E_ + tid] = r;
}

// ---------------------------- launch ----------------------------
extern "C" void kernel_launch(void* const* d_in, const int* in_sizes, int n_in,
                              void* d_out, int out_size, void* d_ws, size_t ws_size,
                              hipStream_t stream) {
  const int*   word2news = (const int*)d_in[0];
  const int*   w2n_len   = (const int*)d_in[1];
  const int*   news_words= (const int*)d_in[2];
  const float* table     = (const float*)d_in[3];
  const float* in_w      = (const float*)d_in[4];
  const float* in_b      = (const float*)d_in[5];
  const float* out_w     = (const float*)d_in[6];
  const float* out_b     = (const float*)d_in[7];
  const float* cw3 = (const float*)d_in[8];  const float* cb3 = (const float*)d_in[9];
  const float* cw4 = (const float*)d_in[10]; const float* cb4 = (const float*)d_in[11];
  const float* cw5 = (const float*)d_in[12]; const float* cb5 = (const float*)d_in[13];
  const float* fc_w = (const float*)d_in[14]; const float* fc_b = (const float*)d_in[15];
  float* out = (float*)d_out;

  char* ws = (char*)d_ws;
  size_t off = 0;
  auto alloc = [&](size_t bytes) -> void* {
    void* p = ws + off;
    off = (off + bytes + 255) & ~(size_t)255;
    return p;
  };
  bf16* we  = (bf16*)alloc((size_t)V_ * E_ * sizeof(bf16));
  bf16* wkp = (bf16*)alloc(4 * 8 * 32 * 16 * sizeof(bf16));
  bf16* wvp = (bf16*)alloc(4 * 8 * 32 * 16 * sizeof(bf16));
  bf16* cp3 = (bf16*)alloc(12 * 8 * 32 * 16 * sizeof(bf16));
  bf16* cp4 = (bf16*)alloc(16 * 8 * 32 * 16 * sizeof(bf16));
  bf16* cp5 = (bf16*)alloc(20 * 8 * 32 * 16 * sizeof(bf16));

  // Pack wk, wv: B(k,n) = in_proj_w[(E+n)*E + k] (wk), [(2E+n)*E + k] (wv)
  pack_b_frags<<<(4 * 8 * 32 + 127) / 128, 128, 0, stream>>>(
      in_w, wkp, 4, 1, E_, E_ * E_, 30, 0);
  pack_b_frags<<<(4 * 8 * 32 + 127) / 128, 128, 0, stream>>>(
      in_w, wvp, 4, 1, E_, 2 * E_ * E_, 30, 0);
  // Pack conv taps: B_dt(e,c) = conv_w[(c*E + e)*ks + dt]; linear ktile = dt*4+kt
  pack_b_frags<<<(12 * 8 * 32 + 127) / 128, 128, 0, stream>>>(
      cw3, cp3, 12, 3, E_ * 3, 0, 7, 1);
  pack_b_frags<<<(16 * 8 * 32 + 127) / 128, 128, 0, stream>>>(
      cw4, cp4, 16, 4, E_ * 4, 0, 7, 1);
  pack_b_frags<<<(20 * 8 * 32 + 127) / 128, 128, 0, stream>>>(
      cw5, cp5, 20, 5, E_ * 5, 0, 7, 1);

  word_attn_kernel<<<V_, 128, 0, stream>>>(word2news, w2n_len, table, in_w, in_b,
                                           out_w, out_b, wkp, wvp, we);
  conv_fc_kernel<<<N_, 128, 0, stream>>>(news_words, we, cp3, cb3, cp4, cb4,
                                         cp5, cb5, fc_w, fc_b, out);
}
